// Q_Mlp_17901423689856
// MI455X (gfx1250) — compile-verified, matrix operand order checked by reading, not run on hardware
//
#include <hip/hip_runtime.h>
#include <hip/hip_bf16.h>

// ---------------------------------------------------------------------------
// HAWQ quantized MLP for MI455X (gfx1250), built around V_WMMA_I32_16X16X64_IU8
// with double-buffered LDS staging via GLOBAL_LOAD_ASYNC_TO_LDS_B128 (ASYNCcnt)
// when the toolchain exposes the builtin.
//   GEMM1: [50176,384]i8 x [1536,384]i8^T -> h (dequant+ReLU, f16) + global max
//   GEMM2: [50176,1536]i8 x [384,1536]i8^T -> y (f32) + global |max|
// ---------------------------------------------------------------------------

typedef __attribute__((ext_vector_type(8))) int v8i;
typedef __attribute__((ext_vector_type(4))) int v4i;
typedef __attribute__((ext_vector_type(4))) float v4f;
typedef __attribute__((ext_vector_type(4))) signed char v4c;
typedef __attribute__((ext_vector_type(8))) _Float16 v8h;
typedef __attribute__((ext_vector_type(8))) signed char v8c;

#define QMAXF 127.0f

// ------------------------- async global->LDS staging -----------------------
#if defined(__has_builtin)
#  if __has_builtin(__builtin_amdgcn_global_load_async_to_lds_b128) && \
      __has_builtin(__builtin_amdgcn_s_wait_asynccnt)
#    define USE_ASYNC_LDS 1
#  endif
#endif
#ifndef USE_ASYNC_LDS
#  define USE_ASYNC_LDS 0
#endif

__device__ __forceinline__ void copy16_g2lds(const signed char* g, unsigned char* l) {
#if USE_ASYNC_LDS
    // global_load_async_to_lds_b128: per-lane 16B, LDS written directly,
    // tracked with ASYNCcnt (no VGPR data staging).
    // Toolchain signature (from diagnostics): (v4i*, v4i*, imm, imm).
    __builtin_amdgcn_global_load_async_to_lds_b128((v4i*)g, (v4i*)l, 0, 0);
#else
    *(uint4*)l = *(const uint4*)g;
#endif
}

__device__ __forceinline__ void async_fence() {
#if USE_ASYNC_LDS
    __builtin_amdgcn_s_wait_asynccnt(0);
#endif
}

__device__ __forceinline__ v8i load_frag_b32x8(const unsigned char* p) {
    const uint4* q = (const uint4*)p;  // LDS, 16B aligned -> ds_load_b128 x2
    uint4 lo = q[0];
    uint4 hi = q[1];
    v8i r;
    r[0] = (int)lo.x; r[1] = (int)lo.y; r[2] = (int)lo.z; r[3] = (int)lo.w;
    r[4] = (int)hi.x; r[5] = (int)hi.y; r[6] = (int)hi.z; r[7] = (int)hi.w;
    return r;
}

// ---------------------------------------------------------------------------
// Tiled int8 GEMM:  C[M,N] = A[M,K] * B[N,K]^T   (both row-major int8)
// Block tile 128x128x64, 256 threads = 8 wave32, wave grid 4(M)x2(N),
// each wave 2x4 WMMA 16x16 accumulators. Double-buffered LDS, 1 barrier/slice.
// MODE 1: h = relu((acc + b_int)*w_sf*act_sf)  -> f16 store + max reduce
// MODE 2: y = (acc + b_int)*w_sf*sf1           -> f32 store + |max| reduce
// ---------------------------------------------------------------------------
template <int MODE>
__global__ __launch_bounds__(256) void gemm_q8_kernel(
    const signed char* __restrict__ A, const signed char* __restrict__ B,
    int K, int N,
    const float* __restrict__ wsf, const float* __restrict__ bint,
    const float* __restrict__ sfsrc,  // MODE1: act_sf ptr; MODE2: hmax ptr
    void* __restrict__ OUT, float* __restrict__ maxslot) {
    __shared__ __align__(16) unsigned char ldsA[2][128 * 64];
    __shared__ __align__(16) unsigned char ldsB[2][128 * 64];
    __shared__ float sred[8];

    const int tid  = threadIdx.x;
    const int lane = tid & 31;
    const int wave = tid >> 5;
    const int wm   = wave & 3;   // 0..3 -> 32-row band
    const int wn   = wave >> 2;  // 0..1 -> 64-col band
    const int lr   = lane & 15;
    const int lh   = lane >> 4;

    const int mBase = blockIdx.x * 128;
    const int nBase = blockIdx.y * 128;

    // cooperative staging: 32B of A + 32B of B per thread per 64-K slice
    const int cr = tid >> 1;        // 0..127 tile row
    const int cc = (tid & 1) * 32;  // 0 / 32 byte offset within 64B row
    const signed char* gA = A + (size_t)(mBase + cr) * K + cc;
    const signed char* gB = B + (size_t)(nBase + cr) * K + cc;

    v8i acc[2][4];
    const v8i vzero = {0, 0, 0, 0, 0, 0, 0, 0};
#pragma unroll
    for (int i = 0; i < 2; ++i)
#pragma unroll
        for (int j = 0; j < 4; ++j) acc[i][j] = vzero;

    auto stage = [&](int kc, int buf) {
        copy16_g2lds(gA + kc,      ldsA[buf] + cr * 64 + cc);
        copy16_g2lds(gA + kc + 16, ldsA[buf] + cr * 64 + cc + 16);
        copy16_g2lds(gB + kc,      ldsB[buf] + cr * 64 + cc);
        copy16_g2lds(gB + kc + 16, ldsB[buf] + cr * 64 + cc + 16);
    };

    stage(0, 0);
    async_fence();
    __syncthreads();

    int cur = 0;
    for (int kc = 0; kc < K; kc += 64) {
        if (kc + 64 < K) {
            stage(kc + 64, cur ^ 1);  // overlaps the WMMAs below
            if (kc + 128 < K) {       // gfx1250 global_prefetch_b8, 2 slices ahead
                __builtin_prefetch((const void*)(gA + kc + 128), 0, 1);
                __builtin_prefetch((const void*)(gB + kc + 128), 0, 1);
            }
        }

        v8i aF[2], bF[4];
#pragma unroll
        for (int i = 0; i < 2; ++i)
            aF[i] = load_frag_b32x8(ldsA[cur] + (wm * 32 + i * 16 + lr) * 64 + lh * 32);
#pragma unroll
        for (int j = 0; j < 4; ++j)
            bF[j] = load_frag_b32x8(ldsB[cur] + (wn * 64 + j * 16 + lr) * 64 + lh * 32);

#pragma unroll
        for (int i = 0; i < 2; ++i)
#pragma unroll
            for (int j = 0; j < 4; ++j)
                acc[i][j] = __builtin_amdgcn_wmma_i32_16x16x64_iu8(
                    true, aF[i], true, bF[j], acc[i][j], false, false);

        async_fence();   // next buffer fully in LDS
        __syncthreads(); // everyone done reading cur before it is rewritten
        cur ^= 1;
    }

    // ----- epilogue: dequant (+bias), activation, store, block max -----
    const float s_act = (MODE == 1) ? sfsrc[0] : sfsrc[0] * (1.0f / QMAXF);
    float lmax = 0.0f;
#pragma unroll
    for (int j = 0; j < 4; ++j) {
        const int col = nBase + wn * 64 + j * 16 + lr;
        const float s  = wsf[col] * s_act;
        const float bb = bint[col];
#pragma unroll
        for (int i = 0; i < 2; ++i) {
            const int rowBase = mBase + wm * 32 + i * 16 + (lh ? 8 : 0);
#pragma unroll
            for (int e = 0; e < 8; ++e) {
                float v = ((float)acc[i][j][e] + bb) * s;
                const size_t idx = (size_t)(rowBase + e) * (size_t)N + (size_t)col;
                if (MODE == 1) {
                    v = fmaxf(v, 0.0f);
                    lmax = fmaxf(lmax, v);
                    ((_Float16*)OUT)[idx] = (_Float16)v;
                } else {
                    lmax = fmaxf(lmax, fabsf(v));
                    ((float*)OUT)[idx] = v;
                }
            }
        }
    }
    // wave32 butterfly, then one atomic per block (values >= 0 -> uint order)
#pragma unroll
    for (int o = 16; o > 0; o >>= 1) lmax = fmaxf(lmax, __shfl_xor(lmax, o));
    if (lane == 0) sred[wave] = lmax;
    __syncthreads();
    if (tid == 0) {
        float m = sred[0];
#pragma unroll
        for (int w = 1; w < 8; ++w) m = fmaxf(m, sred[w]);
        atomicMax((unsigned int*)maxslot, __float_as_uint(m));
    }
}

// ---------------------------------------------------------------------------
// Per-output-channel weight quantization: one block per row.
// ---------------------------------------------------------------------------
__global__ __launch_bounds__(128) void quant_weights_kernel(
    const float* __restrict__ W, signed char* __restrict__ Wq,
    float* __restrict__ wsf, int cols) {
    const int row = blockIdx.x;
    const float* wr = W + (size_t)row * cols;
    float m = 0.0f;
    for (int c = threadIdx.x; c < cols; c += 128) m = fmaxf(m, fabsf(wr[c]));
#pragma unroll
    for (int o = 16; o > 0; o >>= 1) m = fmaxf(m, __shfl_xor(m, o));
    __shared__ float sm[4];
    __shared__ float ssf;
    const int lane = threadIdx.x & 31, wave = threadIdx.x >> 5;
    if (lane == 0) sm[wave] = m;
    __syncthreads();
    if (threadIdx.x == 0) {
        float t = fmaxf(fmaxf(sm[0], sm[1]), fmaxf(sm[2], sm[3]));
        ssf = t * (1.0f / QMAXF);
        wsf[row] = ssf;
    }
    __syncthreads();
    const float inv = 1.0f / ssf;
    signed char* qr = Wq + (size_t)row * cols;
    for (int c = threadIdx.x; c < cols; c += 128) {
        int q = (int)rintf(wr[c] * inv);
        q = max(-128, min(127, q));
        qr[c] = (signed char)q;
    }
}

// b_int = round(b / (w_sf * act_sf));  act_sf = sfsrc[0] * sfmul
__global__ void bias_int_kernel(const float* __restrict__ b,
                                const float* __restrict__ wsf,
                                const float* __restrict__ sfsrc, float sfmul,
                                float* __restrict__ bi, int n) {
    const int i = blockIdx.x * blockDim.x + threadIdx.x;
    if (i < n) bi[i] = rintf(b[i] / (wsf[i] * (sfsrc[0] * sfmul)));
}

// x_int8 = clip(round(x / act_sf)) ; vectorized x4
__global__ void quant_x_kernel(const float* __restrict__ x,
                               signed char* __restrict__ xq,
                               const float* __restrict__ sfsrc, size_t n4) {
    const float inv = 1.0f / sfsrc[0];
    for (size_t i = blockIdx.x * (size_t)blockDim.x + threadIdx.x; i < n4;
         i += (size_t)gridDim.x * blockDim.x) {
        v4f xv = ((const v4f*)x)[i];
        v4c q;
#pragma unroll
        for (int e = 0; e < 4; ++e) {
            int t = (int)rintf(xv[e] * inv);
            q[e] = (signed char)max(-128, min(127, t));
        }
        ((v4c*)xq)[i] = q;
    }
}

// h_int8 = clip(round(h / sf1)),  sf1 = hmax/127 ; vectorized x8
__global__ void requant_h_kernel(const _Float16* __restrict__ h,
                                 signed char* __restrict__ hq,
                                 const float* __restrict__ hmax, size_t n8) {
    const float inv = QMAXF / hmax[0];
    for (size_t i = blockIdx.x * (size_t)blockDim.x + threadIdx.x; i < n8;
         i += (size_t)gridDim.x * blockDim.x) {
        v8h hv = ((const v8h*)h)[i];
        v8c q;
#pragma unroll
        for (int e = 0; e < 8; ++e) {
            int t = (int)rintf((float)hv[e] * inv);
            q[e] = (signed char)max(-128, min(127, t));
        }
        ((v8c*)hq)[i] = q;
    }
}

// y_out = clip(round(y/asf2))*asf2 ; out[4*n4] = asf2 ; vectorized x4
__global__ void finalize_kernel(const float* __restrict__ y,
                                float* __restrict__ out,
                                const float* __restrict__ ymax, size_t n4) {
    const float sf  = ymax[0] * (1.0f / QMAXF);
    const float inv = (sf > 0.0f) ? (1.0f / sf) : 0.0f;
    for (size_t i = blockIdx.x * (size_t)blockDim.x + threadIdx.x; i < n4;
         i += (size_t)gridDim.x * blockDim.x) {
        v4f yv = ((const v4f*)y)[i];
        v4f r;
#pragma unroll
        for (int e = 0; e < 4; ++e) {
            int t = (int)rintf(yv[e] * inv);
            t = max(-128, min(127, t));
            r[e] = (float)t * sf;
        }
        ((v4f*)out)[i] = r;
    }
    if (blockIdx.x == 0 && threadIdx.x == 0) out[n4 * 4] = sf;
}

// ---------------------------------------------------------------------------
extern "C" void kernel_launch(void* const* d_in, const int* in_sizes, int n_in,
                              void* d_out, int out_size, void* d_ws, size_t ws_size,
                              hipStream_t stream) {
    const size_t M = 50176;  // 256 * 196
    const int D = 384, H = 1536;

    const float* x   = (const float*)d_in[0];
    const float* W1  = (const float*)d_in[1];
    const float* b1  = (const float*)d_in[2];
    const float* W2  = (const float*)d_in[3];
    const float* b2  = (const float*)d_in[4];
    const float* asf = (const float*)d_in[5];
    float* out       = (float*)d_out;

    unsigned char* ws = (unsigned char*)d_ws;
    size_t off = 0;
    auto take = [&](size_t bytes) -> unsigned char* {
        unsigned char* p = ws + off;
        off += (bytes + 255) & ~(size_t)255;
        return p;
    };
    float*       scal = (float*)take(256);  // [0]=hmax [1]=ymax
    signed char* xq   = (signed char*)take(M * D);
    signed char* w1q  = (signed char*)take((size_t)H * D);
    signed char* w2q  = (signed char*)take((size_t)D * H);
    float*       wsf1 = (float*)take((size_t)H * 4);
    float*       b1i  = (float*)take((size_t)H * 4);
    float*       wsf2 = (float*)take((size_t)D * 4);
    float*       b2i  = (float*)take((size_t)D * 4);
    _Float16*    hbuf = (_Float16*)take(M * H * 2);
    signed char* hq   = (signed char*)take(M * H);
    float*       ybuf = (float*)take(M * D * 4);

    float* hmax = scal + 0;
    float* ymax = scal + 1;

    (void)hipMemsetAsync(scal, 0, 256, stream);  // zero atomic-max slots

    // weight / activation quantization
    quant_weights_kernel<<<H, 128, 0, stream>>>(W1, w1q, wsf1, D);
    quant_weights_kernel<<<D, 128, 0, stream>>>(W2, w2q, wsf2, H);
    bias_int_kernel<<<(H + 127) / 128, 128, 0, stream>>>(b1, wsf1, asf, 1.0f, b1i, H);
    quant_x_kernel<<<2048, 256, 0, stream>>>(x, xq, asf, M * D / 4);

    // GEMM1 + ReLU + global max(h)
    {
        dim3 grid((unsigned)(M / 128), (unsigned)(H / 128));
        gemm_q8_kernel<1><<<grid, 256, 0, stream>>>(xq, w1q, D, H, wsf1, b1i,
                                                    asf, (void*)hbuf, hmax);
    }
    // sf1-dependent pieces
    bias_int_kernel<<<(D + 127) / 128, 128, 0, stream>>>(b2, wsf2, hmax,
                                                         1.0f / 127.0f, b2i, D);
    requant_h_kernel<<<4096, 256, 0, stream>>>(hbuf, hq, hmax, M * H / 8);

    // GEMM2 + global max(|y|)
    {
        dim3 grid((unsigned)(M / 128), (unsigned)(D / 128));
        gemm_q8_kernel<2><<<grid, 256, 0, stream>>>(hq, w2q, H, D, wsf2, b2i,
                                                    hmax, (void*)ybuf, ymax);
    }
    // final fake-quant + append asf2 scalar
    finalize_kernel<<<2048, 256, 0, stream>>>(ybuf, out, ymax, M * D / 4);

    (void)in_sizes; (void)n_in; (void)out_size; (void)ws_size;
}
